// DeepBiaffineDecoder_69844758168189
// MI455X (gfx1250) — matrix operand order repaired and checked by reading, not compiled, not run on hardware
//
#include <hip/hip_runtime.h>
#include <hip/hip_fp16.h>
#include <math.h>
#include <stdint.h>

// ---- problem dims ----
#define BB   64
#define LLEN 512
#define HH   400
#define L1   513
#define EN   256
#define ELD  128
#define NLB  128
#define MROWS (BB * L1)        // 32832
#define MT    (MROWS / 16)     // 2052
#define KLBL  16640            // 128*128 bilinear + 128 Wl + 128 Wr
#define KBIT  (KLBL / 32)      // 520

typedef __attribute__((ext_vector_type(2)))  float    v2f;
typedef __attribute__((ext_vector_type(8)))  float    v8f;
typedef __attribute__((ext_vector_type(16))) _Float16 v16h;
typedef __attribute__((ext_vector_type(8)))  _Float16 v8h;

static __device__ inline v8f wmma_f32x4(v2f a, v2f b, v8f c) {
  return __builtin_amdgcn_wmma_f32_16x16x4_f32(false, a, false, b, (short)0, c, false, false);
}
static __device__ inline v8f wmma_f16x32(v16h a, v16h b, v8f c) {
  return __builtin_amdgcn_wmma_f32_16x16x32_f16(false, a, false, b, (short)0, c, false, false);
}
static __device__ inline v8f v8f_zero() { v8f z = {0.f,0.f,0.f,0.f,0.f,0.f,0.f,0.f}; return z; }

// ---------------------------------------------------------------------------
// misc: heads1/labels1 (sentinel column prepended), labels1 -> d_out[o2]
// ---------------------------------------------------------------------------
__global__ void k_misc(const int* __restrict__ eh, const int* __restrict__ el,
                       int* __restrict__ heads1, int* __restrict__ labels1,
                       float* __restrict__ out) {
  int idx = blockIdx.x * blockDim.x + threadIdx.x;
  if (idx >= MROWS) return;
  int b = idx / L1, l = idx % L1;
  int h   = (l == 0) ? 0 : eh[b * LLEN + l - 1];
  int lab = (l == 0) ? 0 : el[b * LLEN + l - 1];
  heads1[idx]  = h;
  labels1[idx] = lab;
  out[65536 + idx] = (float)lab;           // labels1 output (B*L1)
}

// ---------------------------------------------------------------------------
// encoder: [M,400] @ [400,768] fused (enh|enm|elh|elm), f32 WMMA, ELU
// block = 128 (4 waves), grid = (2052, 12)
// ---------------------------------------------------------------------------
__global__ void __launch_bounds__(128)
k_encode(const float* __restrict__ mbank, const float* __restrict__ sentinel,
         const float* __restrict__ W0, const float* __restrict__ bi0,
         const float* __restrict__ W1, const float* __restrict__ bi1,
         const float* __restrict__ W2, const float* __restrict__ bi2,
         const float* __restrict__ W3, const float* __restrict__ bi3,
         float* __restrict__ enh, float* __restrict__ enm,
         _Float16* __restrict__ elh16, _Float16* __restrict__ elm16) {
  int lane = threadIdx.x & 31, wave = threadIdx.x >> 5;
  int mtile = blockIdx.x;
  int ntile = blockIdx.y * 4 + wave;           // 0..47
  int ncol  = ntile * 16 + (lane & 15);        // 0..767
  const float *W, *bias; int segw, segbase;
  if (ncol < 256)      { W = W0; bias = bi0; segw = EN;  segbase = 0;   }
  else if (ncol < 512) { W = W1; bias = bi1; segw = EN;  segbase = 256; }
  else if (ncol < 640) { W = W2; bias = bi2; segw = ELD; segbase = 512; }
  else                 { W = W3; bias = bi3; segw = ELD; segbase = 640; }
  int nloc = ncol - segbase;

  int arow = mtile * 16 + (lane & 15);
  int b = arow / L1, l = arow % L1;
  const float* aptr = (l == 0) ? sentinel
                               : (mbank + ((size_t)b * LLEN + (l - 1)) * HH);
  int kh = 2 * (lane >> 4);                    // A lanes<16: K0,1  lanes>=16: K2,3
  v8f acc = v8f_zero();
  for (int k0 = 0; k0 < HH; k0 += 4) {
    v2f a = *(const v2f*)(aptr + k0 + kh);
    v2f bb;
    bb.x = W[(size_t)(k0 + kh) * segw + nloc];
    bb.y = W[(size_t)(k0 + kh + 1) * segw + nloc];
    acc = wmma_f32x4(a, bb, acc);
  }
  float bv = bias[nloc];
#pragma unroll
  for (int r = 0; r < 8; ++r) {
    int mloc = (lane < 16) ? r : (8 + r);
    int row  = mtile * 16 + mloc;
    float x = acc[r] + bv;
    float y = (x > 0.f) ? x : expm1f(x);       // ELU
    if (segbase == 0)        enh[(size_t)row * EN + nloc] = y;
    else if (segbase == 256) enm[(size_t)row * EN + nloc] = y;
    else if (segbase == 512) elh16[(size_t)row * ELD + nloc] = (_Float16)y;
    else                     elm16[(size_t)row * ELD + nloc] = (_Float16)y;
  }
}

// ---------------------------------------------------------------------------
// wd = enh@ba_Wd, we = enm@ba_We (rank-1 bias terms)
// ---------------------------------------------------------------------------
__global__ void k_wdwe(const float* __restrict__ enh, const float* __restrict__ enm,
                       const float* __restrict__ Wd, const float* __restrict__ We,
                       float* __restrict__ wd, float* __restrict__ we) {
  int row = blockIdx.x * blockDim.x + threadIdx.x;
  if (row >= MROWS) return;
  float s1 = 0.f, s2 = 0.f;
  for (int k = 0; k < EN; ++k) {
    s1 += enh[(size_t)row * EN + k] * Wd[k];
    s2 += enm[(size_t)row * EN + k] * We[k];
  }
  wd[row] = s1; we[row] = s2;
}

// ---------------------------------------------------------------------------
// T = enh @ ba_U   [M,256]x[256,256], f32 WMMA. block=128, grid=(2052,4)
// ---------------------------------------------------------------------------
__global__ void __launch_bounds__(128)
k_T(const float* __restrict__ enh, const float* __restrict__ baU, float* __restrict__ T) {
  int lane = threadIdx.x & 31, wave = threadIdx.x >> 5;
  int mtile = blockIdx.x;
  int ntile = blockIdx.y * 4 + wave;
  int n = ntile * 16 + (lane & 15);
  int arow = mtile * 16 + (lane & 15);
  int kh = 2 * (lane >> 4);
  v8f acc = v8f_zero();
  for (int k0 = 0; k0 < EN; k0 += 4) {
    v2f a = *(const v2f*)(enh + (size_t)arow * EN + k0 + kh);
    v2f bb;
    bb.x = baU[(size_t)(k0 + kh) * EN + n];
    bb.y = baU[(size_t)(k0 + kh + 1) * EN + n];
    acc = wmma_f32x4(a, bb, acc);
  }
#pragma unroll
  for (int r = 0; r < 8; ++r) {
    int row = mtile * 16 + ((lane < 16) ? r : (8 + r));
    T[(size_t)row * EN + n] = acc[r];
  }
}

// ---------------------------------------------------------------------------
// S[b,h,m] = T[b,h,:]·enm[b,m,:] + wd[h] + we[m] + ba_b. grid=(33,33,64), 1 wave
// ---------------------------------------------------------------------------
__global__ void __launch_bounds__(32)
k_S(const float* __restrict__ T, const float* __restrict__ enm,
    const float* __restrict__ wd, const float* __restrict__ we,
    const float* __restrict__ ba_b, float* __restrict__ S) {
  int lane = threadIdx.x & 31;
  int ht = blockIdx.x, mt = blockIdx.y, b = blockIdx.z;
  int hl = ht * 16 + (lane & 15); int hcl = (hl < L1) ? hl : (L1 - 1);
  int ml = mt * 16 + (lane & 15); int mcl = (ml < L1) ? ml : (L1 - 1);
  int kh = 2 * (lane >> 4);
  const float* arow = T   + ((size_t)b * L1 + hcl) * EN;
  const float* brow = enm + ((size_t)b * L1 + mcl) * EN;   // B[k][m] = enm[b,m,k]
  v8f acc = v8f_zero();
  for (int k0 = 0; k0 < EN; k0 += 4) {
    v2f a  = *(const v2f*)(arow + k0 + kh);
    v2f bb = *(const v2f*)(brow + k0 + kh);
    acc = wmma_f32x4(a, bb, acc);
  }
  float bias0 = ba_b[0];
  float wem = (ml < L1) ? we[b * L1 + ml] : 0.f;
#pragma unroll
  for (int r = 0; r < 8; ++r) {
    int h = ht * 16 + ((lane < 16) ? r : (8 + r));
    if (h < L1 && ml < L1)
      S[((size_t)b * L1 + h) * L1 + ml] = acc[r] + wd[b * L1 + h] + wem + bias0;
  }
}

// ---------------------------------------------------------------------------
// per-column (fixed m): log-softmax over h + gold gather + masked argmax decode
// ---------------------------------------------------------------------------
__global__ void __launch_bounds__(256)
k_nodecol(const float* __restrict__ S, const int* __restrict__ mask,
          const int* __restrict__ heads1, float* __restrict__ out,
          int* __restrict__ predh, float* __restrict__ nodePart) {
  int lane = threadIdx.x & 31, w = threadIdx.x >> 5;
  int c = blockIdx.x * 8 + w;
  if (c >= MROWS) return;
  int b = c / L1, m = c % L1;
  float maskm = (m == 0) ? 0.f : (float)mask[b * LLEN + m - 1];
  float v[17];
  float mx = -INFINITY, bestv = -INFINITY; int besti = 0;
#pragma unroll
  for (int i = 0; i < 17; ++i) {
    int h = lane + 32 * i;
    if (h < L1) {
      float raw = S[((size_t)b * L1 + h) * L1 + m];
      float maskh = (h == 0) ? 0.f : (float)mask[b * LLEN + h - 1];
      float z = raw + logf(maskh + maskm + 1e-45f);
      v[i] = z; mx = fmaxf(mx, z);
      float d = raw + (1.f - maskh) * (-1e8f);   // mm[h]; mm[m] const per col
      if (h == m) d = -INFINITY;                 // diagonal
      if (d > bestv) { bestv = d; besti = h; }   // first-max tie-break
    } else v[i] = -INFINITY;
  }
  for (int o = 16; o; o >>= 1) mx = fmaxf(mx, __shfl_xor(mx, o, 32));
  float s = 0.f;
#pragma unroll
  for (int i = 0; i < 17; ++i)
    if (lane + 32 * i < L1) s += expf(v[i] - mx);
  for (int o = 16; o; o >>= 1) s += __shfl_xor(s, o, 32);
  float lse = mx + logf(s);
  for (int o = 16; o; o >>= 1) {
    float ov = __shfl_xor(bestv, o, 32);
    int   oi = __shfl_xor(besti, o, 32);
    if (ov > bestv || (ov == bestv && oi < besti)) { bestv = ov; besti = oi; }
  }
  if (lane == 0) {
    predh[c] = besti;
    if (m >= 1) out[b * LLEN + (m - 1)] = (float)besti;   // pred_heads output
    int hg = heads1[c];
    float rawg  = S[((size_t)b * L1 + hg) * L1 + m];
    float maskh = (hg == 0) ? 0.f : (float)mask[b * LLEN + hg - 1];
    float zg = rawg + logf(maskh + maskm + 1e-45f);
    nodePart[c] = (m >= 1) ? (lse - zg) : 0.f;            // contributes -ll
  }
}

// ---------------------------------------------------------------------------
// pack U/Wl/Wr into f16 K-major B matrix: Bf16[k][n], k = d*128+e, then Wl, Wr
// ---------------------------------------------------------------------------
__global__ void k_bf16(const float* __restrict__ U, const float* __restrict__ Wl,
                       const float* __restrict__ Wr, _Float16* __restrict__ Bf16) {
  int idx = blockIdx.x * blockDim.x + threadIdx.x;
  if (idx >= KLBL * NLB) return;
  int k = idx >> 7, n = idx & 127;
  float v;
  if (k < 16384)      { int d = k >> 7, e = k & 127; v = U[((size_t)n * ELD + d) * ELD + e]; }
  else if (k < 16512) { int d = k - 16384; v = Wl[n * ELD + d]; }
  else                { int e = k - 16512; v = Wr[n * ELD + e]; }
  Bf16[idx] = (_Float16)v;
}

// ---------------------------------------------------------------------------
// label bilinear GEMM: score[m,n] = (elh_sel[m] ⊗ elm[m]) · U  (+Wl,Wr via K-aug)
// M=32832, K=16640, N=128. f16 WMMA; A built on the fly; B (U) double-buffered
// in LDS via gfx1250 async LDS copies (ASYNCcnt + workgroup barrier pipeline).
// block=256 (8 waves = 8 label tiles), each wave 8 M-subtiles. grid = 257
// ---------------------------------------------------------------------------
__global__ void __launch_bounds__(256)
k_label(const _Float16* __restrict__ elh16, const _Float16* __restrict__ elm16,
        const _Float16* __restrict__ Bf16, const int* __restrict__ hsel,
        float* __restrict__ scoreOut) {
  __shared__ __align__(16) _Float16 ldsB[2][32][NLB];   // 2 x 8KB K-step buffers
  int tid  = threadIdx.x;
  int lane = tid & 31, w = tid >> 5;
  int n0 = w * 16;
  int mt0 = blockIdx.x * 8;
  int nsub = MT - mt0; if (nsub > 8) nsub = 8;
  const _Float16* selRow[8];
  const _Float16* elmRow[8];
#pragma unroll
  for (int st = 0; st < 8; ++st) {
    int mtile = mt0 + st;
    if (mtile < MT) {
      int grow = mtile * 16 + (lane & 15);
      int b = grow / L1;
      selRow[st] = elh16 + ((size_t)b * L1 + hsel[grow]) * ELD;
      elmRow[st] = elm16 + (size_t)grow * ELD;
    } else { selRow[st] = elh16; elmRow[st] = elm16; }
  }
  v8f acc[8];
#pragma unroll
  for (int st = 0; st < 8; ++st) acc[st] = v8f_zero();
  int off = (lane < 16) ? 0 : 8;   // f16 A-frag K sub-block per half-wave

  // flat-address low 32 bits == LDS byte offset (ISA: LDS aperture truncation)
  uint32_t ldsBase = (uint32_t)(uintptr_t)&ldsB[0][0][0];
  uint64_t gBase   = (uint64_t)(uintptr_t)Bf16;
  // stage K-step `step` (8KB contiguous) into buffer `buf`: 2 async B128/thread
  auto stage = [&](int buf, int step) {
    uint32_t l0 = ldsBase + (uint32_t)buf * 8192u + (uint32_t)tid * 16u;
    uint64_t g0 = gBase + (uint64_t)step * 8192u + (uint64_t)tid * 16u;
    asm volatile("global_load_async_to_lds_b128 %0, %1, off"
                 :: "v"(l0), "v"(g0) : "memory");
    asm volatile("global_load_async_to_lds_b128 %0, %1, off offset:4096"
                 :: "v"(l0), "v"(g0) : "memory");   // INST_OFFSET applies to both sides
  };

  stage(0, 0);                                    // prologue: 2 outstanding
  for (int kb = 0; kb < KBIT; ++kb) {
    int buf = kb & 1;
    if (kb + 1 < KBIT) {
      stage(buf ^ 1, kb + 1);                     // 4 outstanding
      asm volatile("s_wait_asynccnt 0x2" ::: "memory");  // oldest 2 (this buf) done
    } else {
      asm volatile("s_wait_asynccnt 0x0" ::: "memory");
    }
    __syncthreads();                              // all waves' stripes visible

    v16h bfrag;
#pragma unroll
    for (int i = 0; i < 8; ++i) {
      bfrag[i]     = ldsB[buf][off + i][n0 + (lane & 15)];
      bfrag[i + 8] = ldsB[buf][16 + off + i][n0 + (lane & 15)];
    }
    int kbase = kb * 32;
#pragma unroll
    for (int st = 0; st < 8; ++st) {
      if (st >= nsub) break;                      // workgroup-uniform
      v16h afrag;
      if (kb < 512) {                       // bilinear region: A = elh_sel[d]*elm[e]
        int d = kbase >> 7, e0 = kbase & 127;
        _Float16 hd = selRow[st][d];
        v8h lo = *(const v8h*)(elmRow[st] + e0 + off);
        v8h hi = *(const v8h*)(elmRow[st] + e0 + 16 + off);
#pragma unroll
        for (int i = 0; i < 8; ++i) { afrag[i] = hd * lo[i]; afrag[i + 8] = hd * hi[i]; }
      } else if (kb < 516) {                // Wl region: A = elh_sel
        int base = (kb - 512) * 32;
        v8h lo = *(const v8h*)(selRow[st] + base + off);
        v8h hi = *(const v8h*)(selRow[st] + base + 16 + off);
#pragma unroll
        for (int i = 0; i < 8; ++i) { afrag[i] = lo[i]; afrag[i + 8] = hi[i]; }
      } else {                              // Wr region: A = elm
        int base = (kb - 516) * 32;
        v8h lo = *(const v8h*)(elmRow[st] + base + off);
        v8h hi = *(const v8h*)(elmRow[st] + base + 16 + off);
#pragma unroll
        for (int i = 0; i < 8; ++i) { afrag[i] = lo[i]; afrag[i + 8] = hi[i]; }
      }
      acc[st] = wmma_f16x32(afrag, bfrag, acc[st]);
    }
    __syncthreads();                              // protect buf before re-stage
  }
#pragma unroll
  for (int st = 0; st < 8; ++st) {
    if (st >= nsub) break;
    int mtile = mt0 + st;
#pragma unroll
    for (int r = 0; r < 8; ++r) {
      int row = mtile * 16 + ((lane < 16) ? r : (8 + r));
      scoreOut[(size_t)row * NLB + n0 + (lane & 15)] = acc[st][r];
    }
  }
}

// ---------------------------------------------------------------------------
// gold labels: log-softmax over 128 labels, gather, per-row partial loss
// ---------------------------------------------------------------------------
__global__ void __launch_bounds__(256)
k_lblgold(const float* __restrict__ score, const float* __restrict__ lbl_b,
          const int* __restrict__ labels1, float* __restrict__ lblPart) {
  int lane = threadIdx.x & 31, w = threadIdx.x >> 5;
  int c = blockIdx.x * 8 + w;
  if (c >= MROWS) return;
  float v[4]; float mx = -INFINITY;
#pragma unroll
  for (int i = 0; i < 4; ++i) {
    int n = lane + 32 * i;
    v[i] = score[(size_t)c * NLB + n] + lbl_b[n];
    mx = fmaxf(mx, v[i]);
  }
  for (int o = 16; o; o >>= 1) mx = fmaxf(mx, __shfl_xor(mx, o, 32));
  float s = 0.f;
#pragma unroll
  for (int i = 0; i < 4; ++i) s += expf(v[i] - mx);
  for (int o = 16; o; o >>= 1) s += __shfl_xor(s, o, 32);
  float lse = mx + logf(s);
  if (lane == 0) {
    int l = c % L1;
    int lg = labels1[c];
    float zg = score[(size_t)c * NLB + lg] + lbl_b[lg];
    lblPart[c] = (l >= 1) ? (lse - zg) : 0.f;
  }
}

// ---------------------------------------------------------------------------
// predicted labels: max over 128 -> pred_lbl output
// ---------------------------------------------------------------------------
__global__ void __launch_bounds__(256)
k_lblpred(const float* __restrict__ score, const float* __restrict__ lbl_b,
          float* __restrict__ out) {
  int lane = threadIdx.x & 31, w = threadIdx.x >> 5;
  int c = blockIdx.x * 8 + w;
  if (c >= MROWS) return;
  float mx = -INFINITY;
#pragma unroll
  for (int i = 0; i < 4; ++i) {
    int n = lane + 32 * i;
    mx = fmaxf(mx, score[(size_t)c * NLB + n] + lbl_b[n]);
  }
  for (int o = 16; o; o >>= 1) mx = fmaxf(mx, __shfl_xor(mx, o, 32));
  if (lane == 0) {
    int b = c / L1, l = c % L1;
    if (l >= 1) out[32768 + b * LLEN + (l - 1)] = mx;
  }
}

// ---------------------------------------------------------------------------
// deterministic final loss reductions (fixed order, no float atomics)
// ---------------------------------------------------------------------------
__global__ void __launch_bounds__(256)
k_reduce(const float* __restrict__ nodePart, const float* __restrict__ lblPart,
         float* __restrict__ out) {
  __shared__ float s1[256], s2[256];
  float a = 0.f, b = 0.f;
  for (int i = threadIdx.x; i < MROWS; i += 256) { a += nodePart[i]; b += lblPart[i]; }
  s1[threadIdx.x] = a; s2[threadIdx.x] = b;
  __syncthreads();
  for (int st = 128; st; st >>= 1) {
    if ((int)threadIdx.x < st) { s1[threadIdx.x] += s1[threadIdx.x + st]; s2[threadIdx.x] += s2[threadIdx.x + st]; }
    __syncthreads();
  }
  if (threadIdx.x == 0) { out[98368] = s1[0]; out[98369] = s2[0]; }
}

// ---------------------------------------------------------------------------
extern "C" void kernel_launch(void* const* d_in, const int* in_sizes, int n_in,
                              void* d_out, int out_size, void* d_ws, size_t ws_size,
                              hipStream_t stream) {
  const float* mbank     = (const float*)d_in[0];
  const int*   edgeheads = (const int*)d_in[1];
  const int*   edgelabels= (const int*)d_in[2];
  /* d_in[3] corefs: unused by reference math */
  const int*   mask      = (const int*)d_in[4];
  const float* sentinel  = (const float*)d_in[5];
  const float* W_enh = (const float*)d_in[6];  const float* b_enh = (const float*)d_in[7];
  const float* W_enm = (const float*)d_in[8];  const float* b_enm = (const float*)d_in[9];
  const float* W_elh = (const float*)d_in[10]; const float* b_elh = (const float*)d_in[11];
  const float* W_elm = (const float*)d_in[12]; const float* b_elm = (const float*)d_in[13];
  const float* ba_U  = (const float*)d_in[14]; const float* ba_Wd = (const float*)d_in[15];
  const float* ba_We = (const float*)d_in[16]; const float* ba_b  = (const float*)d_in[17];
  const float* lbl_U = (const float*)d_in[18]; const float* lbl_Wl= (const float*)d_in[19];
  const float* lbl_Wr= (const float*)d_in[20]; const float* lbl_b = (const float*)d_in[21];
  float* out = (float*)d_out;

  char* ws = (char*)d_ws;
  size_t off = 0;
  auto wsalloc = [&](size_t bytes) -> char* {
    char* p = ws + off;
    off = (off + bytes + 255) & ~(size_t)255;
    return p;
  };
  float*     enh     = (float*)wsalloc((size_t)MROWS * EN * 4);
  float*     enm     = (float*)wsalloc((size_t)MROWS * EN * 4);
  float*     T       = (float*)wsalloc((size_t)MROWS * EN * 4);
  float*     S       = (float*)wsalloc((size_t)BB * L1 * L1 * 4);
  _Float16*  elh16   = (_Float16*)wsalloc((size_t)MROWS * ELD * 2);
  _Float16*  elm16   = (_Float16*)wsalloc((size_t)MROWS * ELD * 2);
  _Float16*  Bf16    = (_Float16*)wsalloc((size_t)KLBL * NLB * 2);
  float*     wd      = (float*)wsalloc((size_t)MROWS * 4);
  float*     we      = (float*)wsalloc((size_t)MROWS * 4);
  int*       heads1  = (int*)wsalloc((size_t)MROWS * 4);
  int*       predh   = (int*)wsalloc((size_t)MROWS * 4);
  int*       labels1 = (int*)wsalloc((size_t)MROWS * 4);
  float*     nodeP   = (float*)wsalloc((size_t)MROWS * 4);
  float*     lblP    = (float*)wsalloc((size_t)MROWS * 4);
  float*     scoreB  = T;   // T is dead after k_S; reuse for label scores

  // 1) heads/labels with sentinel column; labels1 output
  k_misc<<<(MROWS + 255) / 256, 256, 0, stream>>>(edgeheads, edgelabels, heads1, labels1, out);
  // 2) pack label tensor to f16 (independent)
  k_bf16<<<(KLBL * NLB + 255) / 256, 256, 0, stream>>>(lbl_U, lbl_Wl, lbl_Wr, Bf16);
  // 3) fused encoder (f32 WMMA)
  k_encode<<<dim3(MT, 12), 128, 0, stream>>>(mbank, sentinel,
      W_enh, b_enh, W_enm, b_enm, W_elh, b_elh, W_elm, b_elm,
      enh, enm, elh16, elm16);
  // 4) rank-1 bias vectors
  k_wdwe<<<(MROWS + 255) / 256, 256, 0, stream>>>(enh, enm, ba_Wd, ba_We, wd, we);
  // 5) T = enh @ ba_U (f32 WMMA)
  k_T<<<dim3(MT, 4), 128, 0, stream>>>(enh, ba_U, T);
  // 6) S[b] = T[b] @ enm[b]^T + biases (f32 WMMA)
  k_S<<<dim3(33, 33, BB), 32, 0, stream>>>(T, enm, wd, we, ba_b, S);
  // 7) node log-softmax + gold gather + masked argmax decode
  k_nodecol<<<(MROWS + 7) / 8, 256, 0, stream>>>(S, mask, heads1, out, predh, nodeP);
  // 8) gold label bilinear GEMM (f16 WMMA + async-LDS pipeline) + reduction
  k_label<<<(MT + 7) / 8, 256, 0, stream>>>(elh16, elm16, Bf16, heads1, scoreB);
  k_lblgold<<<(MROWS + 7) / 8, 256, 0, stream>>>(scoreB, lbl_b, labels1, lblP);
  // 9) predicted label bilinear GEMM + max
  k_label<<<(MT + 7) / 8, 256, 0, stream>>>(elh16, elm16, Bf16, predh, scoreB);
  k_lblpred<<<(MROWS + 7) / 8, 256, 0, stream>>>(scoreB, lbl_b, out);
  // 10) deterministic loss sums
  k_reduce<<<1, 256, 0, stream>>>(nodeP, lblP, out);

  (void)in_sizes; (void)n_in; (void)out_size; (void)ws_size;
}
